// QuantumKernelPCA_81003083202740
// MI455X (gfx1250) — matrix operand order repaired and checked by reading, not compiled, not run on hardware
//
#include <hip/hip_runtime.h>

typedef __attribute__((ext_vector_type(16))) __bf16 v16bf;
typedef __attribute__((ext_vector_type(8)))  float  v8f;

#define N_ROWS 2048
#define N_REF  2048
#define FEAT   8192
#define NCOMP  32

#define BM 256
#define BN 128
#define BK 32
#define LDP 40   // padded LDS K-stride in bf16 elems (80 B: 16B-aligned, 20-bank row skew)

// round-to-nearest-even f32 -> bf16, packed pair
__device__ __forceinline__ unsigned int pack2bf(float a, float b) {
  unsigned int ua = __float_as_uint(a);
  unsigned int ub = __float_as_uint(b);
  unsigned int ra = (ua + 0x7fffu + ((ua >> 16) & 1u)) >> 16;
  unsigned int rb = ((ub + 0x7fffu + ((ub >> 16) & 1u)) >> 16) << 16;
  return (ra & 0xffffu) | rb;
}

// CDNA5 async global->LDS copy, 16B per lane, tracked by ASYNCcnt
__device__ __forceinline__ void async_b128(unsigned int ldsoff, const void* g) {
  asm volatile("global_load_async_to_lds_b128 %0, %1, off"
               :: "v"(ldsoff), "v"((unsigned long long)g)
               : "memory");
}
__device__ __forceinline__ void wait_async0() {
  asm volatile("s_wait_asynccnt 0x0" ::: "memory");
}

// ---------------- Kernel 1: center + L2-normalize rows, emit bf16 ----------------
__global__ __launch_bounds__(256) void center_norm_bf16(
    const float* __restrict__ x, const float* __restrict__ xref,
    const float* __restrict__ mean,
    unsigned short* __restrict__ A, unsigned short* __restrict__ B) {
  int row = blockIdx.x;                       // 0..4095
  const float* src;
  unsigned short* dst;
  if (row < N_ROWS) { src = x    + (size_t)row            * FEAT; dst = A + (size_t)row            * FEAT; }
  else              { src = xref + (size_t)(row - N_ROWS) * FEAT; dst = B + (size_t)(row - N_ROWS) * FEAT; }

  int t = threadIdx.x;
  float4 v[8];
  float ss = 0.f;
#pragma unroll
  for (int c = 0; c < 8; ++c) {
    int idx = (c * 256 + t) * 4;
    float4 d = *(const float4*)(src + idx);
    float4 m = *(const float4*)(mean + idx);
    d.x -= m.x; d.y -= m.y; d.z -= m.z; d.w -= m.w;
    v[c] = d;
    ss += d.x * d.x + d.y * d.y + d.z * d.z + d.w * d.w;
  }
  __shared__ float red[256];
  red[t] = ss;
  __syncthreads();
  for (int s = 128; s > 0; s >>= 1) {
    if (t < s) red[t] += red[t + s];
    __syncthreads();
  }
  float scale = 1.0f / (sqrtf(red[0]) + 1e-8f);
#pragma unroll
  for (int c = 0; c < 8; ++c) {
    int idx = (c * 256 + t) * 4;
    uint2 p;
    p.x = pack2bf(v[c].x * scale, v[c].y * scale);
    p.y = pack2bf(v[c].z * scale, v[c].w * scale);
    *(uint2*)(dst + idx) = p;
  }
}

// ---------------- Kernel 2: K = (A * B^T)^2 via WMMA bf16, async-LDS double buffer ----
union FragU { uint4 u[2]; v16bf v; };

__global__ __launch_bounds__(256) void gemm_sq(
    const unsigned short* __restrict__ A, const unsigned short* __restrict__ B,
    float* __restrict__ Kbuf) {
  __shared__ __align__(16) unsigned short As[2][BM][LDP];
  __shared__ __align__(16) unsigned short Bs[2][BN][LDP];

  int t    = threadIdx.x;
  int lane = t & 31;
  int wave = t >> 5;
  int wm   = wave & 3;     // 4 waves along M (64 rows each)
  int wn   = wave >> 2;    // 2 waves along N (64 cols each)
  int bm   = blockIdx.x * BM;
  int bn   = blockIdx.y * BN;
  int half = lane >> 4;    // 0/1
  int l16  = lane & 15;

  v8f acc[4][4] = {};

  // staging assignment: thread t copies 16B chunks; 4 rows-of-64 for A, 2 for B
  int srow = t >> 2;            // 0..63
  int scol = (t & 3) * 8;       // 0,8,16,24 (elements)
  const unsigned short* gA = A + (size_t)(bm + srow) * FEAT + scol;
  const unsigned short* gB = B + (size_t)(bn + srow) * FEAT + scol;
  unsigned int lA = (unsigned int)(unsigned long long)&As[0][srow][scol];
  unsigned int lB = (unsigned int)(unsigned long long)&Bs[0][srow][scol];
  const unsigned int bufA  = BM * LDP * 2;   // 20480 B per A buffer
  const unsigned int bufB  = BN * LDP * 2;   // 10240 B per B buffer
  const unsigned int rstep = 64 * LDP * 2;   // 5120 B per 64-row group

  // prologue: fill buffer 0 with kt=0 tile
#pragma unroll
  for (int i = 0; i < 4; ++i) async_b128(lA + i * rstep, gA + (size_t)i * 64 * FEAT);
#pragma unroll
  for (int i = 0; i < 2; ++i) async_b128(lB + i * rstep, gB + (size_t)i * 64 * FEAT);
  wait_async0();
  __syncthreads();

  int cur = 0;
  for (int kt = 0; kt < FEAT; kt += BK) {
    if (kt + BK < FEAT) {
      int nxt = cur ^ 1;
#pragma unroll
      for (int i = 0; i < 4; ++i)
        async_b128(lA + nxt * bufA + i * rstep, gA + (size_t)i * 64 * FEAT + kt + BK);
#pragma unroll
      for (int i = 0; i < 2; ++i)
        async_b128(lB + nxt * bufB + i * rstep, gB + (size_t)i * 64 * FEAT + kt + BK);
      __builtin_prefetch(gA + kt + 2 * BK, 0, 3);   // hint tile after next into L2
    }

    const unsigned short (*Asc)[LDP] = As[cur];
    const unsigned short (*Bsc)[LDP] = Bs[cur];

    // A fragments (16x32 MxK): lane m=l16; v0-3 <- K=half*8..+7, v4-7 <- K=16+half*8..+7
    FragU a[4];
#pragma unroll
    for (int mf = 0; mf < 4; ++mf) {
      const unsigned short* p = &Asc[wm * 64 + mf * 16 + l16][half * 8];
      a[mf].u[0] = *(const uint4*)p;
      a[mf].u[1] = *(const uint4*)(p + 16);
    }
    // B fragments (32x16 KxN): lane n=l16; v0-3 <- K=half*8..+7, v4-7 <- K=16+half*8..+7
    // load all four into distinct registers so dscnt waits can be partial/overlapped
    FragU b[4];
#pragma unroll
    for (int nf = 0; nf < 4; ++nf) {
      const unsigned short* p = &Bsc[wn * 64 + nf * 16 + l16][half * 8];
      b[nf].u[0] = *(const uint4*)p;
      b[nf].u[1] = *(const uint4*)(p + 16);
    }
#pragma unroll
    for (int nf = 0; nf < 4; ++nf)
#pragma unroll
      for (int mf = 0; mf < 4; ++mf)
        acc[mf][nf] = __builtin_amdgcn_wmma_f32_16x16x32_bf16(
            false, a[mf].v, false, b[nf].v, (short)0, acc[mf][nf], false, false);

    wait_async0();
    __syncthreads();
    cur ^= 1;
  }

  // C/D layout: VGPR v -> M = v + (lane>=16)*8, N = lane%16
#pragma unroll
  for (int mf = 0; mf < 4; ++mf) {
#pragma unroll
    for (int nf = 0; nf < 4; ++nf) {
      int col   = bn + wn * 64 + nf * 16 + l16;
      int rbase = bm + wm * 64 + mf * 16 + half * 8;
#pragma unroll
      for (int v = 0; v < 8; ++v) {
        float g = acc[mf][nf][v];
        Kbuf[(size_t)(rbase + v) * N_REF + col] = g * g;
      }
    }
  }
}

// ---------------- Kernel 3: Out = K @ components^T ----------------
__global__ __launch_bounds__(256) void proj(
    const float* __restrict__ Kbuf, const float* __restrict__ comp,
    float* __restrict__ out) {
  int i = blockIdx.x;
  int t = threadIdx.x;
  __shared__ __align__(16) float krow[N_REF];
#pragma unroll
  for (int c = 0; c < 2; ++c) {
    int idx = (c * 256 + t) * 4;
    *(float4*)&krow[idx] = *(const float4*)&Kbuf[(size_t)i * N_REF + idx];
  }
  __syncthreads();
  int c = t & 31;   // component
  int s = t >> 5;   // slice of 256 j's
  const float* cp = comp + (size_t)c * N_REF + s * 256;
  const float* kp = krow + s * 256;
  float acc = 0.f;
#pragma unroll 8
  for (int j = 0; j < 256; ++j) acc += kp[j] * cp[j];
  __shared__ float part[8][32];
  part[s][c] = acc;
  __syncthreads();
  if (t < 32) {
    float sum = 0.f;
#pragma unroll
    for (int s2 = 0; s2 < 8; ++s2) sum += part[s2][t];
    out[(size_t)i * NCOMP + t] = sum;
  }
}

extern "C" void kernel_launch(void* const* d_in, const int* in_sizes, int n_in,
                              void* d_out, int out_size, void* d_ws, size_t ws_size,
                              hipStream_t stream) {
  (void)in_sizes; (void)n_in; (void)out_size; (void)ws_size;
  const float* x    = (const float*)d_in[0];
  const float* xref = (const float*)d_in[1];
  const float* mean = (const float*)d_in[2];
  const float* comp = (const float*)d_in[3];
  float* out = (float*)d_out;

  char* w = (char*)d_ws;
  unsigned short* A = (unsigned short*)w;                                  // 32 MB
  unsigned short* B = (unsigned short*)(w + (size_t)N_ROWS * FEAT * 2);    // 32 MB
  float* Kbuf = (float*)(w + (size_t)(N_ROWS + N_REF) * FEAT * 2);         // 16 MB

  hipLaunchKernelGGL(center_norm_bf16, dim3(N_ROWS + N_REF), dim3(256), 0, stream,
                     x, xref, mean, A, B);
  hipLaunchKernelGGL(gemm_sq, dim3(N_ROWS / BM, N_REF / BN), dim3(256), 0, stream,
                     A, B, Kbuf);
  hipLaunchKernelGGL(proj, dim3(N_ROWS), dim3(256), 0, stream, Kbuf, comp, out);
}
